// SparseyLayer_20298015441344
// MI455X (gfx1250) — matrix operand order repaired and checked by reading, not compiled, not run on hardware
//
#include <hip/hip_runtime.h>

// ---------------- problem constants (from reference) ----------------
#define B_     32
#define M_     64
#define PM_    144
#define PM1_   145          // PM + zero row
#define PF_    128
#define CMS_   8
#define NEUR_  16
#define OUTW_  128          // CMS*NEUR
#define SPLIT_ 16           // K-split factor for the GEMM (BDP: keep >18MB in flight)

typedef __attribute__((ext_vector_type(16))) _Float16 v16h;
typedef __attribute__((ext_vector_type(8)))  _Float16 v8h;
typedef __attribute__((ext_vector_type(8)))  float    v8f;

union HV { v16h v; v8h h[2]; };

// ---------------- kernel 1: cast x -> f16 (+zero row), row sums ----------------
__global__ void sparsey_prep(const float* __restrict__ x,
                             _Float16* __restrict__ X16,
                             float* __restrict__ rowsum) {
  const int bp = blockIdx.x;              // 0 .. B_*PM1_-1
  const int b  = bp / PM1_;
  const int p  = bp % PM1_;
  const int j  = threadIdx.x;             // 128
  float v = (p < PM_) ? x[((size_t)b * PM_ + p) * PF_ + j] : 0.0f;
  X16[((size_t)b * PM1_ + p) * PF_ + j] = (_Float16)v;

  __shared__ float red[128];
  red[j] = v;
  __syncthreads();
  for (int off = 64; off > 0; off >>= 1) {
    if (j < off) red[j] += red[j + off];
    __syncthreads();
  }
  if (j == 0) rowsum[b * PM1_ + p] = red[0];
}

// ---------------- kernel 2: WMMA GEMM, K-split partials ----------------
// grid = (8 n-tiles, SPLIT_ k-splits, 64 MACs), block = 1 wave (32 threads).
// A (16x32 f16): batch rows gathered from X16 via connection table.
// B (32x16 f16): weight columns, f32 -> f16 converted in flight (non-temporal).
__global__ void sparsey_gemm(const _Float16* __restrict__ X16,
                             const float* __restrict__ W,
                             const int* __restrict__ conns,
                             float* __restrict__ partial,
                             int R) {
  const int m    = blockIdx.z;
  const int nt   = blockIdx.x;
  const int s    = blockIdx.y;
  const int lane = threadIdx.x;
  const int half = lane >> 4;     // which K half of the tile this lane holds
  const int l16  = lane & 15;
  const int n    = nt * 16 + l16; // output column

  const int KT  = R * 4;                          // 32-wide K tiles
  const int per = (KT + SPLIT_ - 1) / SPLIT_;
  const int kt0 = s * per;
  const int kt1 = (kt0 + per < KT) ? (kt0 + per) : KT;

  const size_t Kfloats = (size_t)R * PF_;
  const float* Wm = W + (size_t)m * Kfloats * OUTW_;

  v8f acc0 = {0.f, 0.f, 0.f, 0.f, 0.f, 0.f, 0.f, 0.f};
  v8f acc1 = acc0;

#pragma unroll 2
  for (int kt = kt0; kt < kt1; ++kt) {
    const int c   = kt >> 2;                 // connection index (PF_/32 = 4 tiles per conn)
    const int j0  = (kt & 3) * 32;           // offset within the prev-MAC feature row
    const int row = conns[m * R + c];        // in [0, PM_]; PM_ is the zero row

    // --- A tiles: A 16x32, contiguous k order (this code's choice, same as B; not the ISA table order).
    // lanes 0-15: M=lane, K {0..15} in v[0..15]
    // lanes16-31: M=lane-16, K {16..31}
    HV a0, a1, bm;
    const _Float16* xr0 =
        X16 + ((size_t)l16 * PM1_ + row) * PF_ + j0 + half * 16;
    a0.h[0] = *(const v8h*)(xr0);
    a0.h[1] = *(const v8h*)(xr0 + 8);
    const _Float16* xr1 = xr0 + (size_t)16 * PM1_ * PF_;   // batch rows 16..31
    a1.h[0] = *(const v8h*)(xr1);
    a1.h[1] = *(const v8h*)(xr1 + 8);

    // --- B tile: B 32x16, contiguous k as in A (not ISA table order): lanes 0-15 hold K=0..15 (2 per VGPR),
    // lanes 16-31 hold K=16..31, column = l16.
    const float* wc = Wm + (size_t)(kt * 32 + half * 16) * OUTW_ + n;
#pragma unroll
    for (int q = 0; q < 16; ++q)
      bm.v[q] = (_Float16)__builtin_nontemporal_load(wc + (size_t)q * OUTW_);

    // prefetch next K-tile's weight column region
    __builtin_prefetch(wc + (size_t)32 * OUTW_, 0, 1);

    acc0 = __builtin_amdgcn_wmma_f32_16x16x32_f16(
        false, a0.v, false, bm.v, (short)0, acc0, false, false);
    acc1 = __builtin_amdgcn_wmma_f32_16x16x32_f16(
        false, a1.v, false, bm.v, (short)0, acc1, false, false);
  }

  // C/D layout: VGPR r -> M=r (lanes 0-15) / M=8+r (lanes 16-31), N=l16.
#pragma unroll
  for (int r = 0; r < 8; ++r) {
    const int b0 = r + half * 8;
    partial[(((size_t)s * B_ + b0)      * M_ + m) * OUTW_ + n] = acc0[r];
    partial[(((size_t)s * B_ + b0 + 16) * M_ + m) * OUTW_ + n] = acc1[r];
  }
}

// ---------------- threefry2x32: replicate jax.random.gumbel(key(42)) ----------------
__device__ __forceinline__ unsigned rotl32(unsigned x, int r) {
  return (x << r) | (x >> (32 - r));
}

__device__ float gumbel_at(unsigned idx) {
  const unsigned n = B_ * M_ * CMS_ * NEUR_;   // 262144
  const unsigned halfn = n / 2;
  unsigned x0, x1;
  bool second;
  if (idx < halfn) { x0 = idx;         x1 = idx + halfn; second = false; }
  else             { x0 = idx - halfn; x1 = idx;         second = true;  }
  const unsigned k0 = 0u, k1 = 42u;
  const unsigned k2 = k0 ^ k1 ^ 0x1BD11BDAu;
  x0 += k0; x1 += k1;
  const int ra[4] = {13, 15, 26, 6};
  const int rb[4] = {17, 29, 16, 24};
#pragma unroll
  for (int i = 0; i < 4; ++i) { x0 += x1; x1 = rotl32(x1, ra[i]); x1 ^= x0; }
  x0 += k1; x1 += k2 + 1u;
#pragma unroll
  for (int i = 0; i < 4; ++i) { x0 += x1; x1 = rotl32(x1, rb[i]); x1 ^= x0; }
  x0 += k2; x1 += k0 + 2u;
#pragma unroll
  for (int i = 0; i < 4; ++i) { x0 += x1; x1 = rotl32(x1, ra[i]); x1 ^= x0; }
  x0 += k0; x1 += k1 + 3u;
#pragma unroll
  for (int i = 0; i < 4; ++i) { x0 += x1; x1 = rotl32(x1, rb[i]); x1 ^= x0; }
  x0 += k1; x1 += k2 + 4u;
#pragma unroll
  for (int i = 0; i < 4; ++i) { x0 += x1; x1 = rotl32(x1, ra[i]); x1 ^= x0; }
  x0 += k2; x1 += k0 + 5u;
  const unsigned bits = second ? x1 : x0;
  const float f = __uint_as_float((bits >> 9) | 0x3F800000u) - 1.0f;
  const float tiny = 1.1754943508222875e-38f;
  const float u = fmaxf(tiny, f * (1.0f - tiny) + tiny);
  return -logf(-logf(u));
}

// ---------------- kernel 3: reduce partials + epilogue ----------------
// grid = B_*M_ blocks, 128 threads (16 neurons x 8 CMs).
__global__ void sparsey_post(const float* __restrict__ rowsum,
                             const float* __restrict__ partial,
                             const int* __restrict__ conns,
                             const float* __restrict__ thr_min,
                             const float* __restrict__ thr_max,
                             float* __restrict__ out,
                             int R) {
  const int bm = blockIdx.x;
  const int b  = bm / M_;
  const int m  = bm % M_;
  const int t  = threadIdx.x;     // 0..127
  const int cm = t >> 4;
  const int nr = t & 15;

  // sums = sum over RF of per-prev-MAC row sums (zero row handles padding)
  __shared__ float sred[128];
  float sacc = 0.0f;
  for (int c = t; c < R; c += 128)
    sacc += rowsum[b * PM1_ + conns[m * R + c]];
  sred[t] = sacc;
  __syncthreads();
  for (int off = 64; off > 0; off >>= 1) {
    if (t < off) sred[t] += sred[t + off];
    __syncthreads();
  }
  const float sums = sred[0];

  // raw = sum of K-split partials; h = raw / max(sums, 1)
  float raw = 0.0f;
  for (int s = 0; s < SPLIT_; ++s)
    raw += partial[(((size_t)s * B_ + b) * M_ + m) * OUTW_ + t];
  const float h = raw / fmaxf(sums, 1.0f);

  // fam = mean over CMs of max over neurons
  float hm = h;
#pragma unroll
  for (int d = 8; d >= 1; d >>= 1) hm = fmaxf(hm, __shfl_xor(hm, d, 16));
  __shared__ float hms[CMS_];
  if (nr == 0) hms[cm] = hm;
  __syncthreads();
  float fam = 0.0f;
#pragma unroll
  for (int i = 0; i < CMS_; ++i) fam += hms[i];
  fam *= 0.125f;

  float eta = 2.5f / (1.0f + expf(-28.0f * (fam - 0.5f)));
  if (fam < 0.2f) eta = 0.0f;

  // softmax over the 16 neurons of this CM
  const float lg = eta * h;
  float mx = lg;
#pragma unroll
  for (int d = 8; d >= 1; d >>= 1) mx = fmaxf(mx, __shfl_xor(mx, d, 16));
  const float e = expf(lg - mx);
  float se = e;
#pragma unroll
  for (int d = 8; d >= 1; d >>= 1) se += __shfl_xor(se, d, 16);
  const float prob = e / se;

  // gumbel-perturbed argmax (first index wins ties, matching jnp.argmax)
  const unsigned gidx = (((unsigned)(b * M_ + m)) * CMS_ + cm) * NEUR_ + nr;
  float score = logf(prob + 1e-30f) + gumbel_at(gidx);
  float bs = score;
  int   bi = nr;
#pragma unroll
  for (int d = 8; d >= 1; d >>= 1) {
    const float os = __shfl_xor(bs, d, 16);
    const int   oi = __shfl_xor(bi, d, 16);
    if (os > bs || (os == bs && oi < bi)) { bs = os; bi = oi; }
  }

  const float tmin = thr_min[m];
  const float tmax = thr_max[m];
  const float active = (sums >= tmin && sums <= tmax) ? 1.0f : 0.0f;

  out[((size_t)b * M_ + m) * OUTW_ + t] = (nr == bi) ? active : 0.0f;
}

// ---------------- launch ----------------
extern "C" void kernel_launch(void* const* d_in, const int* in_sizes, int n_in,
                              void* d_out, int out_size, void* d_ws, size_t ws_size,
                              hipStream_t stream) {
  const float* x      = (const float*)d_in[0];
  const float* w      = (const float*)d_in[1];
  const int*   conns  = (const int*)d_in[2];
  const float* thrmin = (const float*)d_in[3];
  const float* thrmax = (const float*)d_in[4];
  float*       out    = (float*)d_out;

  const int R = in_sizes[2] / M_;   // RF width (padded)

  // workspace carve-out (all 256B aligned)
  char* ws = (char*)d_ws;
  const size_t szX16 = ((size_t)B_ * PM1_ * PF_ * sizeof(_Float16) + 255) & ~(size_t)255;
  const size_t szRS  = ((size_t)B_ * PM1_ * sizeof(float) + 255) & ~(size_t)255;
  _Float16* X16    = (_Float16*)(ws);
  float*    rowsum = (float*)(ws + szX16);
  float*    partial= (float*)(ws + szX16 + szRS);
  (void)ws_size; (void)n_in; (void)out_size;

  sparsey_prep<<<B_ * PM1_, 128, 0, stream>>>(x, X16, rowsum);
  sparsey_gemm<<<dim3(OUTW_ / 16, SPLIT_, M_), 32, 0, stream>>>(X16, w, conns, partial, R);
  sparsey_post<<<B_ * M_, 128, 0, stream>>>(rowsum, partial, conns, thrmin, thrmax, out, R);
}
